// MergeCrossScanSequencesTo3D_8572754723427
// MI455X (gfx1250) — matrix operand order repaired and checked by reading, not compiled, not run on hardware
//
#include <hip/hip_runtime.h>
#include <cstddef>

typedef __attribute__((ext_vector_type(16))) _Float16 v16h;
typedef __attribute__((ext_vector_type(8)))  float    v8f;
typedef __attribute__((ext_vector_type(4)))  float    v4f;

#define BATCH  4
#define CO     32
#define NDIR   12
#define CIN    (NDIR * CO)   /* 384 */
#define LTOT   32768
#define NSAMP  (BATCH * LTOT)

// A-fragment (16-bit 16x32, M x K) half-slot -> K mapping (ISA 7.12.2):
// lanes 0-15: v0..v3 hold K=0..7,  v4..v7 hold K=16..23
// lanes16-31: v0..v3 hold K=8..15, v4..v7 hold K=24..31
__device__ __forceinline__ int a_kmap(int h, int lane) {
    return (h & 7) + 8 * (((h >> 3) << 1) + (lane >> 4));
}

// Source pointer for direction-block t: ys[b, dir(t), 0, l_src(i,j,k)]
__device__ __forceinline__ const float* chunk_ptr(
    const float* __restrict__ ys, int b, int t, int ii, int jj, int kk)
{
    int g   = t >> 1;                                   // axis-permutation group
    int dir = ((g >> 1) << 2) + (g & 1) + ((t & 1) << 1);
    int p, q, r;
    switch (g) {
        case 0:  p = ii; q = jj; r = kk; break;         // (i,j,k)
        case 1:  p = ii; q = kk; r = jj; break;         // (i,k,j)
        case 2:  p = jj; q = kk; r = ii; break;         // (j,k,i)
        case 3:  p = jj; q = ii; r = kk; break;         // (j,i,k)
        case 4:  p = kk; q = ii; r = jj; break;         // (k,i,j)
        default: p = kk; q = jj; r = ii; break;         // (k,j,i)
    }
    int lsrc = (p << 10) + (q << 5) + r;
    if (t & 1) lsrc = (LTOT - 1) - lsrc;                // reversed scans
    return ys + ((size_t)(b * NDIR + dir) * CO) * LTOT + lsrc;
}

__global__ __launch_bounds__(256) void merge_gemm_stats_kernel(
    const float* __restrict__ ys,      // (4,12,32,32768)
    const float* __restrict__ conv_w,  // (32,384)
    const float* __restrict__ conv_b,  // (32,)
    float* __restrict__ out,           // (4,32,32768) pre-BN
    float* __restrict__ gsum,          // (32,)
    float* __restrict__ gsumsq)        // (32,)
{
    __shared__ __align__(32) _Float16 sA[24 * 512];      // 24 KiB, A register-images
    __shared__ __align__(32) _Float16 sX[2 * 16 * 512];  // 32 KiB, double-buffered B
    __shared__ float redS[CO];
    __shared__ float redQ[CO];

    const int tid   = threadIdx.x;
    const int lane  = tid & 31;
    const int wave  = tid >> 5;
    const int b     = blockIdx.x >> 7;          // 128 tiles per batch
    const int lBase = (blockIdx.x & 127) << 8;  // 256 consecutive l per block

    // ---- Stage all A (weight) fragments once, fp32 -> f16 ----
    for (int idx = tid; idx < 24 * 512; idx += 256) {
        int frag  = idx >> 9;          // m*12 + t
        int pos   = idx & 511;
        int flane = pos >> 4;
        int h     = pos & 15;
        int m  = frag / NDIR;
        int t  = frag % NDIR;
        int mo = flane & 15;
        int kc = a_kmap(h, flane);
        sA[idx] = (_Float16)conv_w[(m * 16 + mo) * CIN + t * CO + kc];
    }
    if (tid < CO) { redS[tid] = 0.0f; redQ[tid] = 0.0f; }

    // Per-thread spatial coordinates (this thread stages column l = lBase+tid)
    const int l    = lBase + tid;
    const int ii   = l >> 10;
    const int jj   = (l >> 5) & 31;
    const int kk   = l & 31;
    const int sSub = tid >> 4;      // 16-wide N subtile this thread stages
    const int nIn  = tid & 15;      // N within the subtile
    const int hi8  = (lane >> 4) << 3;  // C/D layout: lanes16-31 hold M = r+8

    // Accumulators start at conv bias (WMMA C input), so no epilogue add.
    v8f acc[2][2];
    #pragma unroll
    for (int m = 0; m < 2; ++m)
        #pragma unroll
        for (int r = 0; r < 8; ++r) {
            float bv = conv_b[m * 16 + r + hi8];
            acc[m][0][r] = bv;
            acc[m][1][r] = 0.0f;   // bias added once (via acc[m][0]); acc summed per-n? no:
        }
    // NOTE: acc[m][0] and acc[m][1] are *different* output columns; both need bias.
    #pragma unroll
    for (int m = 0; m < 2; ++m)
        #pragma unroll
        for (int r = 0; r < 8; ++r)
            acc[m][1][r] = acc[m][0][r];

    // ---- software pipeline: prefetch chunk 0 into registers ----
    float pre[CO];
    {
        const float* s0 = chunk_ptr(ys, b, 0, ii, jj, kk);
        #pragma unroll
        for (int kc = 0; kc < CO; ++kc) pre[kc] = s0[(size_t)kc * LTOT];
    }

    __syncthreads();

    #pragma unroll
    for (int t = 0; t < NDIR; ++t) {
        _Float16* xb = &sX[(t & 1) * (16 * 512)];

        // convert + scatter prefetched chunk into B register-image layout
        #pragma unroll
        for (int kc = 0; kc < CO; ++kc) {
            int slane = (((kc >> 3) & 1) << 4) + nIn;   // lanes16-31 hold K=8..15,24..31
            xb[sSub * 512 + slane * 16 + ((kc & 7) + ((kc >> 4) << 3))] = (_Float16)pre[kc];
        }
        __syncthreads();

        // issue gathers for chunk t+1 while WMMAs for chunk t execute
        if (t + 1 < NDIR) {
            const float* sn = chunk_ptr(ys, b, t + 1, ii, jj, kk);
            #pragma unroll
            for (int kc = 0; kc < CO; ++kc) pre[kc] = sn[(size_t)kc * LTOT];
        }

        // each wave owns 2 N-subtiles x 2 M-tiles
        #pragma unroll
        for (int s2 = 0; s2 < 2; ++s2) {
            int s = wave * 2 + s2;
            v16h bfrag = *(const v16h*)&xb[s * 512 + lane * 16];
            #pragma unroll
            for (int m = 0; m < 2; ++m) {
                v16h afrag = *(const v16h*)&sA[(m * NDIR + t) * 512 + lane * 16];
                acc[m][s2] = __builtin_amdgcn_wmma_f32_16x16x32_f16(
                    false, afrag, false, bfrag, (short)0, acc[m][s2], false, false);
            }
        }
    }

    // ---- epilogue: store pre-BN; shuffle-reduce per-channel sum / sumsq ----
    #pragma unroll
    for (int m = 0; m < 2; ++m) {
        #pragma unroll
        for (int r = 0; r < 8; ++r) {
            int o = m * 16 + r + hi8;
            float ps = 0.0f, pq = 0.0f;
            #pragma unroll
            for (int s2 = 0; s2 < 2; ++s2) {
                int nl  = (wave * 2 + s2) * 16 + (lane & 15);
                float v = acc[m][s2][r];
                out[((size_t)b * CO + o) * LTOT + lBase + nl] = v;
                ps += v; pq += v * v;
            }
            // reduce across the 16 lanes of each half-wave (same o per half)
            #pragma unroll
            for (int off = 1; off < 16; off <<= 1) {
                ps += __shfl_xor(ps, off, 32);
                pq += __shfl_xor(pq, off, 32);
            }
            if ((lane & 15) == 0) {
                atomicAdd(&redS[o], ps);
                atomicAdd(&redQ[o], pq);
            }
        }
    }
    __syncthreads();
    if (tid < CO) {
        atomicAdd(&gsum[tid],   redS[tid]);
        atomicAdd(&gsumsq[tid], redQ[tid]);
    }
}

__global__ __launch_bounds__(256) void bn_finalize_kernel(
    float* __restrict__ out,
    const float* __restrict__ gsum, const float* __restrict__ gsumsq,
    const float* __restrict__ gamma, const float* __restrict__ beta)
{
    __shared__ float sMean[CO], sInv[CO], sG[CO], sB[CO];
    if (threadIdx.x < CO) {
        float mean = gsum[threadIdx.x] * (1.0f / NSAMP);
        float var  = gsumsq[threadIdx.x] * (1.0f / NSAMP) - mean * mean;
        sMean[threadIdx.x] = mean;
        sInv[threadIdx.x]  = rsqrtf(var + 1e-5f);
        sG[threadIdx.x] = gamma[threadIdx.x];
        sB[threadIdx.x] = beta[threadIdx.x];
    }
    __syncthreads();
    size_t idx4 = ((size_t)blockIdx.x * blockDim.x + threadIdx.x) * 4;
    int o = (int)((idx4 / LTOT) & (CO - 1));   // LTOT % 4 == 0: float4 never crosses o
    float mean = sMean[o], inv = sInv[o], ga = sG[o], be = sB[o];
    v4f v = *(const v4f*)(out + idx4);
    #pragma unroll
    for (int e = 0; e < 4; ++e) v[e] = (v[e] - mean) * inv * ga + be;
    *(v4f*)(out + idx4) = v;
}

extern "C" void kernel_launch(void* const* d_in, const int* in_sizes, int n_in,
                              void* d_out, int out_size, void* d_ws, size_t ws_size,
                              hipStream_t stream) {
    const float* ys     = (const float*)d_in[0];
    const float* conv_w = (const float*)d_in[1];
    const float* conv_b = (const float*)d_in[2];
    const float* gamma  = (const float*)d_in[3];
    const float* beta   = (const float*)d_in[4];
    float* out    = (float*)d_out;
    float* gsum   = (float*)d_ws;
    float* gsumsq = gsum + CO;

    hipMemsetAsync(d_ws, 0, 2 * CO * sizeof(float), stream);

    merge_gemm_stats_kernel<<<BATCH * (LTOT / 256), 256, 0, stream>>>(
        ys, conv_w, conv_b, out, gsum, gsumsq);

    int total = BATCH * CO * LTOT;                 // 4,194,304 floats
    bn_finalize_kernel<<<(total / 4) / 256, 256, 0, stream>>>(
        out, gsum, gsumsq, gamma, beta);
}
